// CorrNet3D_35424890257898
// MI455X (gfx1250) — compile-verified, matrix operand order checked by reading, not run on hardware
//
#include <hip/hip_runtime.h>
#include <hip/hip_fp16.h>

typedef _Float16 f16;
typedef __attribute__((ext_vector_type(16))) _Float16 v16h;
typedef __attribute__((ext_vector_type(8)))  float    v8f;

#define Bv   8
#define Nv   1024
#define KNN  20

// ---------------------------------------------------------------------------
// WMMA fragment load from a row-major 16x32 f16 tile in LDS (row pitch 32).
// ISA 16-bit A-matrix layout: lane L<16 holds row L, K={0..7,16..23};
// lane L+16 holds row L, K={8..15,24..31}.  Two 16B reads per fragment.
// ---------------------------------------------------------------------------
__device__ inline v16h ld_frag(const f16* sm, int row0, int lane) {
    int r = lane & 15;
    int h = (lane >> 4) & 1;
    const f16* p = sm + (row0 + r) * 32;
    v16h f;
    ((uint4*)&f)[0] = *(const uint4*)(p + h * 8);
    ((uint4*)&f)[1] = *(const uint4*)(p + 16 + h * 8);
    return f;
}

// ---------------------------------------------------------------------------
// C[M,N] = act( A[M,K] * B[N,K]^T + bias ), A/B f16 row-major, C f32.
// M%64==0, N%64==0, K%32==0. Batched via blockIdx.z (strides in elements;
// stride 0 => shared operand).  act: 0 none, 1 relu, 2 leaky(0.2).
//
// Tile staging uses the CDNA5 async copy engine (global_load_async_to_lds,
// ASYNCcnt) with double-buffered LDS so the next K-tile's loads overlap the
// current tile's v_wmma_f32_16x16x32_f16 work.
// ---------------------------------------------------------------------------
__global__ __launch_bounds__(128)
void gemm_nt_f16(const f16* __restrict__ A, const f16* __restrict__ B,
                 float* __restrict__ C, const float* __restrict__ bias,
                 int M, int N, int K, int lda, int ldb, int ldc,
                 long strideA, long strideB, long strideC, int act)
{
    __shared__ __align__(16) f16 Ash[2][64 * 32];
    __shared__ __align__(16) f16 Bsh[2][64 * 32];

    const int bz = blockIdx.z;
    A += (long)bz * strideA;
    B += (long)bz * strideB;
    C += (long)bz * strideC;

    const int tid   = threadIdx.x;
    const int lane  = tid & 31;
    const int wave  = tid >> 5;   // 0..3
    const int wm    = wave >> 1;  // row half of 64x64 tile
    const int wn    = wave & 1;   // col half
    const int tileM = blockIdx.y * 64;
    const int tileN = blockIdx.x * 64;

    // LDS byte addresses (low 32 bits of a flat LDS pointer = LDS offset).
    const unsigned aBase = (unsigned)(unsigned long)(&Ash[0][0]);
    const unsigned bBase = (unsigned)(unsigned long)(&Bsh[0][0]);

    // Async-copy one 64x32 f16 tile of each operand into LDS buffer `buf`.
    // 256 16B chunks per operand; 128 threads issue 2 chunks each.
    auto issue_tiles = [&](int buf, int k0) {
#pragma unroll
        for (int i = 0; i < 2; ++i) {
            int chunk = tid + i * 128;
            int r  = chunk >> 2;            // 0..63
            int c8 = (chunk & 3) * 8;       // f16 col: 0,8,16,24
            unsigned la = aBase + (unsigned)buf * 4096u + (unsigned)(r * 32 + c8) * 2u;
            unsigned lb = bBase + (unsigned)buf * 4096u + (unsigned)(r * 32 + c8) * 2u;
            const f16* ga = A + (long)(tileM + r) * lda + k0 + c8;
            const f16* gb = B + (long)(tileN + r) * ldb + k0 + c8;
            asm volatile("global_load_async_to_lds_b128 %0, %1, off"
                         :: "v"(la), "v"(ga) : "memory");
            asm volatile("global_load_async_to_lds_b128 %0, %1, off"
                         :: "v"(lb), "v"(gb) : "memory");
        }
    };

    v8f acc[2][2] = {};

    issue_tiles(0, 0);
    asm volatile("s_wait_asynccnt 0x0" ::: "memory");
    __syncthreads();

    for (int k0 = 0; k0 < K; k0 += 32) {
        const int cur  = (k0 >> 5) & 1;
        const bool more = (k0 + 32) < K;
        if (more) issue_tiles(cur ^ 1, k0 + 32);   // overlap with WMMA below

        v16h af[2], bf[2];
#pragma unroll
        for (int i = 0; i < 2; ++i) af[i] = ld_frag(Ash[cur], wm * 32 + i * 16, lane);
#pragma unroll
        for (int j = 0; j < 2; ++j) bf[j] = ld_frag(Bsh[cur], wn * 32 + j * 16, lane);

#pragma unroll
        for (int i = 0; i < 2; ++i)
#pragma unroll
            for (int j = 0; j < 2; ++j)
                acc[i][j] = __builtin_amdgcn_wmma_f32_16x16x32_f16(
                    false, af[i], false, bf[j], (short)0, acc[i][j], false, false);

        if (more) asm volatile("s_wait_asynccnt 0x0" ::: "memory");
        __syncthreads();
    }

    // Store: C/D layout — VGPR r, lane L : M = r + 8*(L>>4), N = L&15.
    const int half = lane >> 4, ln = lane & 15;
#pragma unroll
    for (int i = 0; i < 2; ++i)
#pragma unroll
        for (int j = 0; j < 2; ++j) {
            int row0 = tileM + wm * 32 + i * 16 + half * 8;
            int col  = tileN + wn * 32 + j * 16 + ln;
            float bvv = bias ? bias[col] : 0.f;
#pragma unroll
            for (int r = 0; r < 8; ++r) {
                float v = acc[i][j][r] + bvv;
                if (act == 1)      v = v > 0.f ? v : 0.f;
                else if (act == 2) v = v > 0.f ? v : 0.2f * v;
                C[(long)(row0 + r) * ldc + col] = v;
            }
        }
}

// ---------------------------------------------------------------------------
// Utility kernels
// ---------------------------------------------------------------------------
__global__ void fill_f16(f16* p, long n) {
    long i = (long)blockIdx.x * blockDim.x + threadIdx.x;
    if (i < n) p[i] = (f16)0.f;
}

// rows x cin (f32) -> rows x cout (f16), zero-padded columns
__global__ void cvt_pad_f16(const float* __restrict__ in, f16* __restrict__ out,
                            long rows, int cin, int cout) {
    long i = (long)blockIdx.x * blockDim.x + threadIdx.x;
    if (i >= rows * (long)cout) return;
    int  c = (int)(i % cout);
    long r = i / cout;
    out[i] = (c < cin) ? (f16)in[r * cin + c] : (f16)0.f;
}

__global__ void row_sumsq(const float* __restrict__ x, float* __restrict__ out,
                          long rows, int C, int ld) {
    long i = (long)blockIdx.x * blockDim.x + threadIdx.x;
    if (i >= rows) return;
    const float* p = x + i * (long)ld;
    float s = 0.f;
    for (int c = 0; c < C; ++c) { float v = p[c]; s += v * v; }
    out[i] = s;
}

// Per point: neg_d[m] = 2*inner[n,m] - xx[n] - xx[m]; select top-20 indices.
__global__ __launch_bounds__(256)
void topk20(const float* __restrict__ inner, const float* __restrict__ xx,
            int* __restrict__ idx) {
    long row = blockIdx.x;             // b*N + n
    long b   = row / Nv;
    int  n   = (int)(row % Nv);
    const float* irow = inner + row * (long)Nv;
    const float* xb   = xx + b * (long)Nv;
    const float  xn   = xb[n];
    int t = threadIdx.x;
    float val[4];
#pragma unroll
    for (int i = 0; i < 4; ++i) {
        int m = t + i * 256;
        val[i] = 2.f * irow[m] - xn - xb[m];
    }
    __shared__ float sv[256];
    __shared__ int   si[256];
    int* out = idx + row * KNN;
    for (int kk = 0; kk < KNN; ++kk) {
        float best = val[0]; int bi = 0;
#pragma unroll
        for (int i = 1; i < 4; ++i) if (val[i] > best) { best = val[i]; bi = i; }
        sv[t] = best; si[t] = t + bi * 256;
        __syncthreads();
        for (int s = 128; s > 0; s >>= 1) {
            if (t < s) {
                if (sv[t + s] > sv[t] ||
                    (sv[t + s] == sv[t] && si[t + s] < si[t])) {
                    sv[t] = sv[t + s]; si[t] = si[t + s];
                }
            }
            __syncthreads();
        }
        int win = si[0];
        if (t == 0) out[kk] = win;
        if ((win & 255) == t) val[win >> 8] = -3.4e38f;
        __syncthreads();
    }
}

// edge[(b*N+n)*KNN+kk][c]     = feat[b,nb,c] - feat[b,n,c]
// edge[(b*N+n)*KNN+kk][Cin+c] = feat[b,n,c]
__global__ void edge_build(const float* __restrict__ feat, int ldf,
                           const int* __restrict__ idx, f16* __restrict__ edge,
                           int Cin, int W, long total) {
    long i = (long)blockIdx.x * blockDim.x + threadIdx.x;
    if (i >= total) return;
    int  c  = (int)(i % Cin); long t = i / Cin;
    int  kk = (int)(t % KNN); t /= KNN;
    int  n  = (int)(t % Nv);
    long b  = t / Nv;
    const float* fb = feat + b * (long)Nv * ldf;
    int nb = idx[(b * Nv + n) * (long)KNN + kk];
    float ctr = fb[(long)n  * ldf + c];
    float nbv = fb[(long)nb * ldf + c];
    f16* er = edge + ((b * Nv + n) * (long)KNN + kk) * W;
    er[c]       = (f16)(nbv - ctr);
    er[Cin + c] = (f16)ctr;
}

// Deterministic BN stats, stage 1: per-chunk partial sums (no atomics).
__global__ void bn_stats(const float* __restrict__ y, float* __restrict__ partial,
                         int O, long rowsPerChunk, long rows) {
    int  o  = threadIdx.x;               // blockDim.x == O
    long r0 = (long)blockIdx.x * rowsPerChunk;
    long r1 = r0 + rowsPerChunk; if (r1 > rows) r1 = rows;
    float s = 0.f, ss = 0.f;
    for (long r = r0; r < r1; ++r) {
        float v = y[r * O + o]; s += v; ss += v * v;
    }
    partial[(long)blockIdx.x * 2 * O + o]     = s;
    partial[(long)blockIdx.x * 2 * O + O + o] = ss;
}

// Stage 2: fixed-order reduce -> mv[0:O]=mean, mv[O:2O]=var (biased).
__global__ void bn_reduce(const float* __restrict__ partial, float* __restrict__ mv,
                          int O, int chunks, float cnt) {
    int o = blockIdx.x * blockDim.x + threadIdx.x;
    if (o >= O) return;
    float s = 0.f, ss = 0.f;
    for (int c = 0; c < chunks; ++c) {
        s  += partial[(long)c * 2 * O + o];
        ss += partial[(long)c * 2 * O + O + o];
    }
    float mu = s / cnt;
    mv[o]     = mu;
    mv[O + o] = ss / cnt - mu * mu;
}

// BN + LeakyReLU(0.2) + max over k; writes f32 + f16 (both offset, pitch ldo).
__global__ void bn_lrelu_max(const float* __restrict__ y, const float* __restrict__ mv,
                             const float* __restrict__ g, const float* __restrict__ bb,
                             float* __restrict__ outf, f16* __restrict__ outh,
                             int O, int ldo, long total) {
    long i = (long)blockIdx.x * blockDim.x + threadIdx.x;
    if (i >= total) return;
    int  o = (int)(i % O); long t = i / O;   // t = b*N+n
    float mu = mv[o], var = mv[O + o];
    float sc = g[o] * rsqrtf(var + 1e-5f);
    float be = bb[o] - mu * sc;
    const float* yr = y + t * (long)KNN * O + o;
    float m = -3.4e38f;
    for (int kk = 0; kk < KNN; ++kk) {
        float v = yr[(long)kk * O] * sc + be;
        v = v > 0.f ? v : 0.2f * v;
        m = v > m ? v : m;
    }
    outf[t * (long)ldo + o] = m;
    outh[t * (long)ldo + o] = (f16)m;
}

// GroupNorm over [B,N,C] point-major, in place; optional f16 mirror.
// act: 1 relu, 2 leaky(0.2).
__global__ __launch_bounds__(256)
void group_norm(float* __restrict__ x, const float* __restrict__ g,
                const float* __restrict__ bgn, f16* __restrict__ outh,
                int C, int groups, int act) {
    int grp = blockIdx.x, b = blockIdx.y;
    int Cg = C / groups;
    long base = (long)b * Nv * C + grp * Cg;
    int  t = threadIdx.x;
    long cnt = (long)Nv * Cg;
    float s = 0.f, ss = 0.f;
    for (long i = t; i < cnt; i += 256) {
        long n = i / Cg; int c = (int)(i % Cg);
        float v = x[base + n * C + c]; s += v; ss += v * v;
    }
    __shared__ float rs[256], rss[256];
    rs[t] = s; rss[t] = ss; __syncthreads();
    for (int sh = 128; sh > 0; sh >>= 1) {
        if (t < sh) { rs[t] += rs[t + sh]; rss[t] += rss[t + sh]; }
        __syncthreads();
    }
    float mu = rs[0] / (float)cnt;
    float var = rss[0] / (float)cnt - mu * mu;
    float rstd = rsqrtf(var + 1e-5f);
    for (long i = t; i < cnt; i += 256) {
        long n = i / Cg; int c = (int)(i % Cg);
        long off = base + n * C + c;
        float v = (x[off] - mu) * rstd * g[grp * Cg + c] + bgn[grp * Cg + c];
        if (act == 1)      v = v > 0.f ? v : 0.f;
        else if (act == 2) v = v > 0.f ? v : 0.2f * v;
        x[off] = v;
        if (outh) outh[off] = (f16)v;
    }
}

// glob[b][0:C]=max_n, glob[b][C:2C]=mean_n of y[b,n,c]
__global__ void pool_max_mean(const float* __restrict__ y, float* __restrict__ glob,
                              int C) {
    int i = blockIdx.x * blockDim.x + threadIdx.x;
    if (i >= Bv * C) return;
    int c = i % C, b = i / C;
    const float* p = y + (long)b * Nv * C + c;
    float m = -3.4e38f, s = 0.f;
    for (int n = 0; n < Nv; ++n) { float v = p[(long)n * C]; m = v > m ? v : m; s += v; }
    glob[(long)b * 2 * C + c]     = m;
    glob[(long)b * 2 * C + C + c] = s / (float)Nv;
}

// h[b,n,c] = c<512 ? localh[b,n,c] : (f16)glob[b][c-512]
__global__ void build_h(const f16* __restrict__ localh, const float* __restrict__ glob,
                        f16* __restrict__ h, long total) {
    long i = (long)blockIdx.x * blockDim.x + threadIdx.x;
    if (i >= total) return;
    int  c = (int)(i & 1023); long t = i >> 10;   // t = b*N+n
    long b = t / Nv;
    h[i] = (c < 512) ? localh[t * 512 + c] : (f16)glob[b * 512 + (c - 512)];
}

// simh = f16( 1/(na[n]+nc[m]-2*inner + 1e-6) )
__global__ void sim_build(const float* __restrict__ inner, const float* __restrict__ na,
                          const float* __restrict__ nc, f16* __restrict__ simh,
                          long total) {
    long i = (long)blockIdx.x * blockDim.x + threadIdx.x;
    if (i >= total) return;
    int  m = (int)(i % Nv); long t = i / Nv;
    int  n = (int)(t % Nv); long b = t / Nv;
    float d = na[b * Nv + n] + nc[b * Nv + m] - 2.f * inner[i];
    simh[i] = (f16)(1.f / (d + 1e-6f));
}

// Row standardize (ddof=1), write f16:  out = (x - mu)/(sd + 1e-6)
__global__ __launch_bounds__(256)
void row_norm_f16(const float* __restrict__ x, f16* __restrict__ out, int C) {
    long row = blockIdx.x;
    const float* p = x + row * (long)C;
    int t = threadIdx.x;
    float s = 0.f, ss = 0.f;
    for (int c = t; c < C; c += 256) { float v = p[c]; s += v; ss += v * v; }
    __shared__ float rs[256], rss[256];
    rs[t] = s; rss[t] = ss; __syncthreads();
    for (int sh = 128; sh > 0; sh >>= 1) {
        if (t < sh) { rs[t] += rs[t + sh]; rss[t] += rss[t + sh]; }
        __syncthreads();
    }
    float mu  = rs[0] / (float)C;
    float var = (rss[0] - (float)C * mu * mu) / (float)(C - 1);
    float sd  = sqrtf(fmaxf(var, 0.f));
    float inv = 1.f / (sd + 1e-6f);
    for (int c = t; c < C; c += 256)
        out[row * (long)C + c] = (f16)((p[c] - mu) * inv);
}

// Column standardize (ddof=1) + *8 + column softmax, in place.
// Thread owns one column, loops rows (coalesced block-wide row reads).
__global__ __launch_bounds__(256)
void col_norm_softmax(float* __restrict__ x) {
    int col = blockIdx.x * 256 + threadIdx.x;
    int b   = blockIdx.y;
    float* p = x + (long)b * Nv * Nv + col;
    float s = 0.f, ss = 0.f, mx = -3.4e38f;
    for (int r = 0; r < Nv; ++r) {
        float v = p[(long)r * Nv];
        s += v; ss += v * v; mx = v > mx ? v : mx;
    }
    float mu  = s / (float)Nv;
    float var = (ss - (float)Nv * mu * mu) / (float)(Nv - 1);
    float sd  = sqrtf(fmaxf(var, 0.f));
    float inv = 8.f / (sd + 1e-6f);          // grad_mul8 forward = x*8
    float zmax = (mx - mu) * inv;
    float se = 0.f;
    for (int r = 0; r < Nv; ++r) {
        long o = (long)r * Nv;
        float e = __expf((p[o] - mu) * inv - zmax);
        p[o] = e; se += e;
    }
    float rinv = 1.f / se;
    for (int r = 0; r < Nv; ++r) p[(long)r * Nv] *= rinv;
}

// ---------------------------------------------------------------------------
// Host orchestration
// ---------------------------------------------------------------------------
static inline long cdiv(long a, long b) { return (a + b - 1) / b; }

extern "C" void kernel_launch(void* const* d_in, const int* in_sizes, int n_in,
                              void* d_out, int out_size, void* d_ws, size_t ws_size,
                              hipStream_t stream) {
    (void)in_sizes; (void)n_in; (void)out_size; (void)ws_size;

    const float* xyz1 = (const float*)d_in[0];
    const float* xyz2 = (const float*)d_in[1];
    // params in dict order:
    const float* conv1_w = (const float*)d_in[2];
    const float* bn1_g = (const float*)d_in[3];  const float* bn1_b = (const float*)d_in[4];
    const float* conv2_w = (const float*)d_in[5];
    const float* bn2_g = (const float*)d_in[6];  const float* bn2_b = (const float*)d_in[7];
    const float* conv3_w = (const float*)d_in[8];
    const float* bn3_g = (const float*)d_in[9];  const float* bn3_b = (const float*)d_in[10];
    const float* conv4_w = (const float*)d_in[11];
    const float* bn4_g = (const float*)d_in[12]; const float* bn4_b = (const float*)d_in[13];
    const float* conv5_w = (const float*)d_in[14];
    const float* gn5_g = (const float*)d_in[15]; const float* gn5_b = (const float*)d_in[16];
    const float* mlp1_w = (const float*)d_in[17]; const float* mlp1_b = (const float*)d_in[18];
    const float* gn6_g = (const float*)d_in[19]; const float* gn6_b = (const float*)d_in[20];
    const float* mlp2_w = (const float*)d_in[21]; const float* mlp2_b = (const float*)d_in[22];
    const float* gn7_g = (const float*)d_in[23]; const float* gn7_b = (const float*)d_in[24];
    const float* mlp3_w = (const float*)d_in[25]; const float* mlp3_b = (const float*)d_in[26];
    const float* gn8_g = (const float*)d_in[27]; const float* gn8_b = (const float*)d_in[28];
    const float* ds1_w = (const float*)d_in[29];
    const float* ds2_w = (const float*)d_in[30];

    // ---- bump allocator over workspace
    char* wp = (char*)d_ws;
    auto alloc = [&](size_t bytes) -> void* {
        void* r = wp; wp += (bytes + 255) & ~(size_t)255; return r;
    };

    // f16 weights (zero-padded K where needed)
    f16* w1h  = (f16*)alloc((size_t)64 * 32 * 2);
    f16* w2h  = (f16*)alloc((size_t)64 * 128 * 2);
    f16* w3h  = (f16*)alloc((size_t)128 * 128 * 2);
    f16* w4h  = (f16*)alloc((size_t)256 * 256 * 2);
    f16* w5h  = (f16*)alloc((size_t)256 * 512 * 2);
    f16* wm1h = (f16*)alloc((size_t)512 * 1024 * 2);
    f16* wm2h = (f16*)alloc((size_t)256 * 512 * 2);
    f16* wm3h = (f16*)alloc((size_t)128 * 256 * 2);
    f16* wd1h = (f16*)alloc((size_t)1152 * 1024 * 2);
    f16* wd2h = (f16*)alloc((size_t)1024 * 1152 * 2);

    const long BN   = (long)Bv * Nv;           // 8192 points
    const long ROWS = BN * KNN;                // 163840 edge rows
    f16*   x0h    = (f16*)alloc((size_t)BN * 32 * 2);
    float* localf = (float*)alloc((size_t)BN * 512 * 4);
    f16*   localh = (f16*)alloc((size_t)BN * 512 * 2);
    float* xx     = (float*)alloc((size_t)BN * 4);
    float* inner  = (float*)alloc((size_t)BN * Nv * 4);
    int*   idx    = (int*)alloc((size_t)BN * KNN * 4);
    f16*   edgeh  = (f16*)alloc((size_t)ROWS * 256 * 2);
    float* ybuf   = (float*)alloc((size_t)ROWS * 256 * 4);
    float* part   = (float*)alloc((size_t)64 * 2 * 256 * 4);
    float* mv     = (float*)alloc((size_t)2 * 256 * 4);
    float* y5     = (float*)alloc((size_t)BN * 256 * 4);
    float* glob   = (float*)alloc((size_t)Bv * 512 * 4);
    f16*   hin    = (f16*)alloc((size_t)BN * 1024 * 2);
    float* h1     = (float*)alloc((size_t)BN * 512 * 4);
    f16*   h1h    = (f16*)alloc((size_t)BN * 512 * 2);
    float* h2     = (float*)alloc((size_t)BN * 256 * 4);
    f16*   h2h    = (f16*)alloc((size_t)BN * 256 * 2);
    float* h3     = (float*)alloc((size_t)BN * 128 * 4);
    f16*   f1h    = (f16*)alloc((size_t)BN * 128 * 2);
    f16*   f2h    = (f16*)alloc((size_t)BN * 128 * 2);
    float* na     = (float*)alloc((size_t)BN * 4);
    float* nc     = (float*)alloc((size_t)BN * 4);
    f16*   simh   = (f16*)alloc((size_t)BN * Nv * 2);
    float* X1     = (float*)alloc((size_t)BN * 1152 * 4);
    f16*   X1n    = (f16*)alloc((size_t)BN * 1152 * 2);

    auto cvt = [&](const float* in, f16* out, long rows, int cin, int cout) {
        long tot = rows * cout;
        cvt_pad_f16<<<dim3((unsigned)cdiv(tot, 256)), 256, 0, stream>>>(in, out, rows, cin, cout);
    };
    auto gemm = [&](const f16* A, const f16* B, float* C, const float* bias,
                    int M, int Nn, int K, int lda, int ldb, int ldc,
                    long sA, long sB, long sC, int act) {
        dim3 grid((unsigned)(Nn / 64), (unsigned)(M / 64), Bv);
        gemm_nt_f16<<<grid, 128, 0, stream>>>(A, B, C, bias, M, Nn, K,
                                              lda, ldb, ldc, sA, sB, sC, act);
    };

    // ---- one-time weight conversion (deterministic each call)
    cvt(conv1_w, w1h, 64, 6, 32);
    cvt(conv2_w, w2h, 64, 128, 128);
    cvt(conv3_w, w3h, 128, 128, 128);
    cvt(conv4_w, w4h, 256, 256, 256);
    cvt(conv5_w, w5h, 256, 512, 512);
    cvt(mlp1_w, wm1h, 512, 1024, 1024);
    cvt(mlp2_w, wm2h, 256, 512, 512);
    cvt(mlp3_w, wm3h, 128, 256, 256);
    cvt(ds1_w, wd1h, 1152, 1024, 1024);
    cvt(ds2_w, wd2h, 1024, 1152, 1152);

    // ---- one edge-conv layer (shared helper)
    auto edge_layer = [&](const float* featf, const f16* feath, int ldf, int Cin, int Kpad,
                          const f16* wh, int O, const float* bg, const float* bb,
                          int coff) {
        // Gram: inner[b] = feath * feath^T   (K = Kpad)
        row_sumsq<<<dim3((unsigned)cdiv(BN, 256)), 256, 0, stream>>>(featf, xx, BN, Cin, ldf);
        gemm(feath, feath, inner, nullptr, Nv, Nv, Kpad, ldf, ldf, Nv,
             (long)Nv * ldf, (long)Nv * ldf, (long)Nv * Nv, 0);
        topk20<<<dim3((unsigned)BN), 256, 0, stream>>>(inner, xx, idx);
        int W = 2 * Cin < 32 ? 32 : 2 * Cin;
        if (W > 2 * Cin)
            fill_f16<<<dim3((unsigned)cdiv(ROWS * W, 256)), 256, 0, stream>>>(edgeh, ROWS * (long)W);
        long tot = ROWS * (long)Cin;
        edge_build<<<dim3((unsigned)cdiv(tot, 256)), 256, 0, stream>>>(featf, ldf, idx, edgeh, Cin, W, tot);
        // y = edge * w^T  : M = N*KNN per batch, N = O, K = W
        gemm(edgeh, wh, ybuf, nullptr, Nv * KNN, O, W, W, W, O,
             (long)Nv * KNN * W, 0, (long)Nv * KNN * O, 0);
        // BN stats (deterministic 2-stage) + lrelu + max over k
        bn_stats<<<dim3(64), O, 0, stream>>>(ybuf, part, O, ROWS / 64, ROWS);
        bn_reduce<<<dim3((unsigned)cdiv(O, 64)), 64, 0, stream>>>(part, mv, O, 64, (float)ROWS);
        long tot2 = BN * (long)O;
        bn_lrelu_max<<<dim3((unsigned)cdiv(tot2, 256)), 256, 0, stream>>>(
            ybuf, mv, bg, bb, localf + coff, localh + coff, O, 512, tot2);
    };

    // ---- full encoder
    auto encoder = [&](const float* xyz, f16* fh_out, float* nrm_out) {
        cvt(xyz, x0h, BN, 3, 32);
        edge_layer(xyz,            x0h,          3,   3,   32, w1h,  64, bn1_g, bn1_b, 0);
        edge_layer(localf + 0,     localh + 0,   512, 64,  64, w2h,  64, bn2_g, bn2_b, 64);
        edge_layer(localf + 64,    localh + 64,  512, 64,  64, w3h, 128, bn3_g, bn3_b, 128);
        edge_layer(localf + 128,   localh + 128, 512, 128, 128, w4h, 256, bn4_g, bn4_b, 256);

        // conv5: [B,N,512] -> [B,N,256]; GN(32) + leaky
        gemm(localh, w5h, y5, nullptr, Nv, 256, 512, 512, 512, 256,
             (long)Nv * 512, 0, (long)Nv * 256, 0);
        group_norm<<<dim3(32, Bv), 256, 0, stream>>>(y5, gn5_g, gn5_b, (f16*)nullptr, 256, 32, 2);
        pool_max_mean<<<dim3((unsigned)cdiv(Bv * 256, 256)), 256, 0, stream>>>(y5, glob, 256);
        build_h<<<dim3((unsigned)cdiv(BN * 1024, 256)), 256, 0, stream>>>(localh, glob, hin, BN * 1024);

        // mlp1 -> gn6(relu) -> mlp2 -> gn7(relu) -> mlp3 -> gn8(relu)
        gemm(hin, wm1h, h1, mlp1_b, Nv, 512, 1024, 1024, 1024, 512,
             (long)Nv * 1024, 0, (long)Nv * 512, 0);
        group_norm<<<dim3(32, Bv), 256, 0, stream>>>(h1, gn6_g, gn6_b, h1h, 512, 32, 1);
        gemm(h1h, wm2h, h2, mlp2_b, Nv, 256, 512, 512, 512, 256,
             (long)Nv * 512, 0, (long)Nv * 256, 0);
        group_norm<<<dim3(32, Bv), 256, 0, stream>>>(h2, gn7_g, gn7_b, h2h, 256, 32, 1);
        gemm(h2h, wm3h, h3, mlp3_b, Nv, 128, 256, 256, 256, 128,
             (long)Nv * 256, 0, (long)Nv * 128, 0);
        group_norm<<<dim3(32, Bv), 256, 0, stream>>>(h3, gn8_g, gn8_b, fh_out, 128, 32, 1);
        row_sumsq<<<dim3((unsigned)cdiv(BN, 256)), 256, 0, stream>>>(h3, nrm_out, BN, 128, 128);
    };

    encoder(xyz1, f1h, na);
    encoder(xyz2, f2h, nc);

    // ---- correspondence head
    // inner = f1 * f2^T ; sim = 1/(P + 1e-6)
    gemm(f1h, f2h, inner, nullptr, Nv, Nv, 128, 128, 128, Nv,
         (long)Nv * 128, (long)Nv * 128, (long)Nv * Nv, 0);
    sim_build<<<dim3((unsigned)cdiv(BN * Nv, 256)), 256, 0, stream>>>(inner, na, nc, simh, BN * Nv);

    // ds1: relu(sim * ds1_w^T) -> row standardize (over 1152 channels, ddof=1)
    gemm(simh, wd1h, X1, nullptr, Nv, 1152, 1024, 1024, 1024, 1152,
         (long)Nv * Nv, 0, (long)Nv * 1152, 1);
    row_norm_f16<<<dim3((unsigned)BN), 256, 0, stream>>>(X1, X1n, 1152);

    // ds2 -> d_out (row-major [B, n1, n2]); then column standardize + *8 + softmax
    float* out = (float*)d_out;
    gemm(X1n, wd2h, out, nullptr, Nv, 1024, 1152, 1152, 1152, 1024,
         (long)Nv * 1152, 0, (long)Nv * Nv, 0);
    col_norm_softmax<<<dim3(4, Bv), 256, 0, stream>>>(out);
}
